// RichContactAttention_81578608820462
// MI455X (gfx1250) — compile-verified, matrix operand order checked by reading, not run on hardware
//
#include <hip/hip_runtime.h>
#include <hip/hip_bf16.h>

// ---------------------------------------------------------------------------
// RichContactAttention for MI455X (gfx1250, wave32, WMMA 16x16x32 f16)
//
// Roofline: ~75 GFLOP, ~232 MB HBM traffic (obs once + outputs). Weights
// (<0.5MB) pre-packed to f16 WMMA B-fragments in workspace -> L2 resident.
// All GEMMs via v_wmma_f32_16x16x32_f16 with f32 accumulation; softmax / LN /
// head MLP / reductions in f32 VALU. Batch-mean reductions are fixed-order
// tree reductions (deterministic across graph replays).
// ---------------------------------------------------------------------------

typedef __attribute__((ext_vector_type(16))) _Float16 v16h;
typedef __attribute__((ext_vector_type(8)))  _Float16 v8h;
typedef __attribute__((ext_vector_type(8)))  float    v8f;

#define B_TOT 131072

// packed-weight offsets (in halves). Tile = 512 halves (32 lanes x 16).
#define O_W1 0         // 10 kt x 16 nt  = 160 tiles (K=308->320, N=256)
#define O_W2 81920     //  8 kt x 32 nt  = 256 tiles (K=256,      N=512)
#define O_WR 212992    //  1 kt x  4 nt  (K=11->32, N=64)
#define O_WQ 215040    //  2 kt x  4 nt
#define O_WK 219136
#define O_WV 223232
#define O_WO 227328    // end 231424 halves = 462848 B

// workspace byte offsets
#define WS_RAW   524288u
#define WS_LW   1048576u
#define WS_PART 1572864u   // 256 partials + 2 scalars

__device__ __forceinline__ v8f wmma16(v16h a, v16h b, v8f c) {
  return __builtin_amdgcn_wmma_f32_16x16x32_f16(false, a, false, b, (short)0, c,
                                                false, false);
}

// load a 16-half A-fragment from f16 LDS row (k layout: j<8 -> k0+j ; j>=8 -> k0+16+j-8)
__device__ __forceinline__ v16h ld_frag16(const _Float16* p) {
  v8h lo = *(const v8h*)(p);
  v8h hi = *(const v8h*)(p + 16);
  v16h r;
#pragma unroll
  for (int j = 0; j < 8; ++j) { r[j] = lo[j]; r[j + 8] = hi[j]; }
  return r;
}

// ---------------------------------------------------------------------------
// Kernel 0: pack all weights into f16 WMMA B-fragment layout.
// B-frag lane layout (16-bit operands, wave32):
//   lane l: n = nt*16 + (l&15), g = l>>4
//   half j: k = kt*32 + ((j<8)? j : j+8) + g*8
// ---------------------------------------------------------------------------
__global__ void pack_weights(const float* __restrict__ W1, const float* __restrict__ W2,
                             const float* __restrict__ Wr, const float* __restrict__ Win,
                             const float* __restrict__ Wout, _Float16* __restrict__ pack) {
  int t = blockIdx.x * blockDim.x + threadIdx.x;
  int tile = t >> 5, lane = t & 31;
  if (tile >= 452) return;
  const float* src; int ldK, Kv, Nv, Nt, rel, base;
  if (tile < 160)      { rel = tile;       Nt = 16; src = W1;         ldK = 308; Kv = 308; Nv = 256; base = O_W1; }
  else if (tile < 416) { rel = tile - 160; Nt = 32; src = W2;         ldK = 256; Kv = 256; Nv = 512; base = O_W2; }
  else if (tile < 420) { rel = tile - 416; Nt = 4;  src = Wr;         ldK = 11;  Kv = 11;  Nv = 64;  base = O_WR; }
  else if (tile < 428) { rel = tile - 420; Nt = 4;  src = Win;        ldK = 64;  Kv = 64;  Nv = 64;  base = O_WQ; }
  else if (tile < 436) { rel = tile - 428; Nt = 4;  src = Win + 4096; ldK = 64;  Kv = 64;  Nv = 64;  base = O_WK; }
  else if (tile < 444) { rel = tile - 436; Nt = 4;  src = Win + 8192; ldK = 64;  Kv = 64;  Nv = 64;  base = O_WV; }
  else                 { rel = tile - 444; Nt = 4;  src = Wout;       ldK = 64;  Kv = 64;  Nv = 64;  base = O_WO; }
  int kt = rel / Nt, nt = rel % Nt;
  int n = nt * 16 + (lane & 15), g = lane >> 4;
  _Float16* dst = pack + base + rel * 512 + lane * 16;
#pragma unroll
  for (int j = 0; j < 16; ++j) {
    int k = kt * 32 + ((j < 8) ? j : j + 8) + g * 8;
    float v = (k < Kv && n < Nv) ? src[n * ldK + k] : 0.0f;
    dst[j] = (_Float16)v;
  }
}

// ---------------------------------------------------------------------------
// Kernel 1: main fused pipeline. 64 threads = 2 waves, 16 batch rows / wave.
// Per-wave LDS region (bytes):
//   h    : 16x256 f16 =  8192   @ 0
//   ctok : 16x512 f16 = 16384   @ 8192
//   remb : 16x64  f32 =  4096   @ 24576
//   q    : 16x64  f32 =  4096   @ 28672
//   k    :128x64  f32 = 32768   @ 32768
//   v    :128x64  f32 = 32768   @ 65536
//   attnP: 16x4x8 f32 =  2048   @ 98304
//   x    : 16x64  f32 =  4096   @ 100352
//   ctx  : 16x64  f16 =  2048   @ 104448   (total 106496)
// ---------------------------------------------------------------------------
#define WAVE_LDS 106496

__global__ void __launch_bounds__(64, 1)
rca_main(const float* __restrict__ obs, const _Float16* __restrict__ pack,
         const float* __restrict__ b1, const float* __restrict__ b2,
         const float* __restrict__ br, const float* __restrict__ bin_,
         const float* __restrict__ bout, const float* __restrict__ gamma,
         const float* __restrict__ beta, const float* __restrict__ Wh1,
         const float* __restrict__ bh1, const float* __restrict__ Wh2,
         const float* __restrict__ bh2, float* __restrict__ out,
         float* __restrict__ raw_ws) {
  extern __shared__ unsigned char smem[];
  const int tid = threadIdx.x;
  const int wv = tid >> 5;
  const int lane = tid & 31;
  const int g = lane >> 4;
  const int ln = lane & 15;

  unsigned char* wb = smem + wv * WAVE_LDS;
  _Float16* h_lds   = (_Float16*)(wb);
  _Float16* ct_lds  = (_Float16*)(wb + 8192);
  float*    remb    = (float*)(wb + 24576);
  float*    q_lds   = (float*)(wb + 28672);
  float*    k_lds   = (float*)(wb + 32768);
  float*    v_lds   = (float*)(wb + 65536);
  float*    attnP   = (float*)(wb + 98304);
  float*    x_lds   = (float*)(wb + 100352);
  _Float16* ctx_lds = (_Float16*)(wb + 104448);

  const int myRow = blockIdx.x * 32 + wv * 16 + ln;
  const float* orow = obs + (size_t)myRow * 376;

  // ---- Stage 1: contact gather -> A-frags; GEMM1 (16x320 x 320x256) + ReLU
  // contact col c -> obs col: c<224 ? c+45 : c+68 (tiles 0..6 fully <224)
  v16h afr[10];
#pragma unroll
  for (int kt = 0; kt < 9; ++kt) {
    int c0 = ((kt < 7) ? 45 : 68) + kt * 32 + g * 8;
    v16h f;
#pragma unroll
    for (int j = 0; j < 8; ++j) {
      f[j]     = (_Float16)orow[c0 + j];
      f[j + 8] = (_Float16)orow[c0 + 16 + j];
    }
    afr[kt] = f;
  }
  { // kt = 9: pad k>=308 with zero
    v16h f;
#pragma unroll
    for (int j = 0; j < 8; ++j) {
      int k0 = 288 + g * 8 + j;
      int k1 = 288 + 16 + g * 8 + j;
      f[j]     = (_Float16)((k0 < 308) ? orow[k0 + 68] : 0.0f);
      f[j + 8] = (_Float16)((k1 < 308) ? orow[k1 + 68] : 0.0f);
    }
    afr[9] = f;
  }
  for (int nt = 0; nt < 16; ++nt) {
    v8f acc = {};
#pragma unroll
    for (int kt = 0; kt < 10; ++kt) {
      v16h bfr = *(const v16h*)(pack + O_W1 + (kt * 16 + nt) * 512 + lane * 16);
      acc = wmma16(afr[kt], bfr, acc);
    }
    float bias = b1[nt * 16 + ln];
#pragma unroll
    for (int r = 0; r < 8; ++r) {
      float hv = fmaxf(acc[r] + bias, 0.0f);
      h_lds[(r + g * 8) * 256 + nt * 16 + ln] = (_Float16)hv;
    }
  }
  __syncthreads();

  // ---- Stage 2: GEMM2 (16x256 x 256x512) -> c_tokens
  v16h hfr[8];
#pragma unroll
  for (int kt = 0; kt < 8; ++kt)
    hfr[kt] = ld_frag16(h_lds + ln * 256 + kt * 32 + g * 8);
  for (int nt = 0; nt < 32; ++nt) {
    v8f acc = {};
#pragma unroll
    for (int kt = 0; kt < 8; ++kt) {
      v16h bfr = *(const v16h*)(pack + O_W2 + (kt * 32 + nt) * 512 + lane * 16);
      acc = wmma16(hfr[kt], bfr, acc);
    }
    float bias = b2[nt * 16 + ln];
#pragma unroll
    for (int r = 0; r < 8; ++r)
      ct_lds[(r + g * 8) * 512 + nt * 16 + ln] = (_Float16)(acc[r] + bias);
  }
  __syncthreads();

  // ---- Stage 3: r_emb = root @ Wr.T + br  (root cols: 0..4, 22..27)
  {
    v16h rfr;
#pragma unroll
    for (int j = 0; j < 16; ++j) {
      int k = ((j < 8) ? j : j + 8) + g * 8;
      float v = 0.0f;
      if (k < 11) v = orow[(k < 5) ? k : (k + 17)];
      rfr[j] = (_Float16)v;
    }
    for (int nt = 0; nt < 4; ++nt) {
      v8f acc = {};
      v16h bfr = *(const v16h*)(pack + O_WR + nt * 512 + lane * 16);
      acc = wmma16(rfr, bfr, acc);
      float bias = br[nt * 16 + ln];
#pragma unroll
      for (int r = 0; r < 8; ++r)
        remb[(r + g * 8) * 64 + nt * 16 + ln] = acc[r] + bias;
    }
  }
  __syncthreads();

  // ---- Stage 3b: q = r_emb @ Wq.T + bq
  {
    v16h refr[2];
#pragma unroll
    for (int kt = 0; kt < 2; ++kt) {
      v16h f;
#pragma unroll
      for (int j = 0; j < 8; ++j) {
        f[j]     = (_Float16)remb[ln * 64 + kt * 32 + g * 8 + j];
        f[j + 8] = (_Float16)remb[ln * 64 + kt * 32 + 16 + g * 8 + j];
      }
      refr[kt] = f;
    }
    for (int nt = 0; nt < 4; ++nt) {
      v8f acc = {};
#pragma unroll
      for (int kt = 0; kt < 2; ++kt) {
        v16h bfr = *(const v16h*)(pack + O_WQ + (kt * 4 + nt) * 512 + lane * 16);
        acc = wmma16(refr[kt], bfr, acc);
      }
      float bias = bin_[nt * 16 + ln];
#pragma unroll
      for (int r = 0; r < 8; ++r)
        q_lds[(r + g * 8) * 64 + nt * 16 + ln] = acc[r] + bias;
    }
  }

  // ---- Stage 4: k,v projections over 128 token-rows (8 M-tiles)
  for (int mt = 0; mt < 8; ++mt) {
    int tr = mt * 16 + ln;              // token-row = (wave-local row)*8 + token
    int crow = tr >> 3, ctk = tr & 7;
    v16h cfr[2];
#pragma unroll
    for (int kt = 0; kt < 2; ++kt)
      cfr[kt] = ld_frag16(ct_lds + crow * 512 + ctk * 64 + kt * 32 + g * 8);
    for (int nt = 0; nt < 4; ++nt) {
      v8f acck = {}, accv = {};
#pragma unroll
      for (int kt = 0; kt < 2; ++kt) {
        v16h bk = *(const v16h*)(pack + O_WK + (kt * 4 + nt) * 512 + lane * 16);
        v16h bv = *(const v16h*)(pack + O_WV + (kt * 4 + nt) * 512 + lane * 16);
        acck = wmma16(cfr[kt], bk, acck);
        accv = wmma16(cfr[kt], bv, accv);
      }
      float biask = bin_[64 + nt * 16 + ln];
      float biasv = bin_[128 + nt * 16 + ln];
#pragma unroll
      for (int r = 0; r < 8; ++r) {
        k_lds[(mt * 16 + r + g * 8) * 64 + nt * 16 + ln] = acck[r] + biask;
        v_lds[(mt * 16 + r + g * 8) * 64 + nt * 16 + ln] = accv[r] + biasv;
      }
    }
  }
  __syncthreads();

  // ---- Stage 5: attention. Each lane: (row=ln, head=g) and (row=ln, head=g+2)
#pragma unroll
  for (int p = 0; p < 2; ++p) {
    int head = g + p * 2;
    float qv[16];
#pragma unroll
    for (int d = 0; d < 16; ++d) qv[d] = q_lds[ln * 64 + head * 16 + d];
    float sc[8]; float mx = -3.4e38f;
#pragma unroll
    for (int t = 0; t < 8; ++t) {
      float s = 0.0f;
#pragma unroll
      for (int d = 0; d < 16; ++d)
        s += qv[d] * k_lds[(ln * 8 + t) * 64 + head * 16 + d];
      s *= 0.25f;                       // 1/sqrt(HD=16)
      sc[t] = s; mx = fmaxf(mx, s);
    }
    float sum = 0.0f;
#pragma unroll
    for (int t = 0; t < 8; ++t) { sc[t] = __expf(sc[t] - mx); sum += sc[t]; }
    float inv = 1.0f / sum;
#pragma unroll
    for (int t = 0; t < 8; ++t) { sc[t] *= inv; attnP[ln * 32 + head * 8 + t] = sc[t]; }
#pragma unroll
    for (int d = 0; d < 16; ++d) {
      float c = 0.0f;
#pragma unroll
      for (int t = 0; t < 8; ++t)
        c += sc[t] * v_lds[(ln * 8 + t) * 64 + head * 16 + d];
      ctx_lds[ln * 64 + head * 16 + d] = (_Float16)c;
    }
  }
  __syncthreads();

  // ---- Stage 6: attn_out = ctx @ Wout.T + bout ; x = attn_out + r_emb
  {
    v16h xfr[2];
#pragma unroll
    for (int kt = 0; kt < 2; ++kt)
      xfr[kt] = ld_frag16(ctx_lds + ln * 64 + kt * 32 + g * 8);
    for (int nt = 0; nt < 4; ++nt) {
      v8f acc = {};
#pragma unroll
      for (int kt = 0; kt < 2; ++kt) {
        v16h bfr = *(const v16h*)(pack + O_WO + (kt * 4 + nt) * 512 + lane * 16);
        acc = wmma16(xfr[kt], bfr, acc);
      }
      float bias = bout[nt * 16 + ln];
#pragma unroll
      for (int r = 0; r < 8; ++r) {
        int m = r + g * 8;
        x_lds[m * 64 + nt * 16 + ln] = acc[r] + bias + remb[m * 64 + nt * 16 + ln];
      }
    }
  }
  __syncthreads();

  // ---- Stage 7: LayerNorm + head MLP + entropy (row = ln; lanes>=16 redundant)
  {
    float sum = 0.0f, sumsq = 0.0f;
    for (int i = 0; i < 64; ++i) {
      float xv = x_lds[ln * 64 + i];
      sum += xv; sumsq += xv * xv;
    }
    float mu = sum * (1.0f / 64.0f);
    float var = sumsq * (1.0f / 64.0f) - mu * mu;
    float inv = rsqrtf(var + 1e-5f);
    float hacc[32];
#pragma unroll
    for (int j = 0; j < 32; ++j) hacc[j] = 0.0f;
    for (int i = 0; i < 64; ++i) {
      float cv = gamma[i] * (x_lds[ln * 64 + i] - mu) * inv + beta[i];
      if (lane < 16) out[(size_t)myRow * 64 + i] = cv;
#pragma unroll
      for (int j = 0; j < 32; ++j) hacc[j] += cv * Wh1[j * 64 + i];
    }
    float rawv = bh2[0];
#pragma unroll
    for (int j = 0; j < 32; ++j)
      rawv += fmaxf(hacc[j] + bh1[j], 0.0f) * Wh2[j];
    if (lane < 16) {
      raw_ws[myRow] = rawv;
      float ent = 0.0f;
#pragma unroll
      for (int t = 0; t < 8; ++t) {
        float aw = 0.25f * (attnP[ln * 32 + t] + attnP[ln * 32 + 8 + t] +
                            attnP[ln * 32 + 16 + t] + attnP[ln * 32 + 24 + t]);
        aw = fmaxf(aw, 1e-8f);
        ent -= aw * __logf(aw);
      }
      out[(size_t)B_TOT * 65 + myRow] = ent;
    }
  }
}

// ---------------------------------------------------------------------------
// Deterministic batch-mean reductions + loss_weight epilogue
// ---------------------------------------------------------------------------
__global__ void reduce_partial(const float* __restrict__ src, float* __restrict__ part, int n) {
  __shared__ float sm[256];
  int chunk = n / gridDim.x;
  int base = blockIdx.x * chunk;
  float s = 0.0f;
  for (int i = threadIdx.x; i < chunk; i += blockDim.x) s += src[base + i];
  sm[threadIdx.x] = s;
  __syncthreads();
  for (int st = 128; st > 0; st >>= 1) {
    if (threadIdx.x < st) sm[threadIdx.x] += sm[threadIdx.x + st];
    __syncthreads();
  }
  if (threadIdx.x == 0) part[blockIdx.x] = sm[0];
}

__global__ void reduce_final(const float* __restrict__ part, float* __restrict__ scal, float scale) {
  __shared__ float sm[256];
  sm[threadIdx.x] = part[threadIdx.x];
  __syncthreads();
  for (int st = 128; st > 0; st >>= 1) {
    if (threadIdx.x < st) sm[threadIdx.x] += sm[threadIdx.x + st];
    __syncthreads();
  }
  if (threadIdx.x == 0) scal[0] = sm[0] * scale;
}

__global__ void lw_kernel(const float* __restrict__ raw, const float* __restrict__ meanRaw,
                          float* __restrict__ lw, int n) {
  int i = blockIdx.x * blockDim.x + threadIdx.x;
  if (i >= n) return;
  float c = raw[i] - meanRaw[0];
  lw[i] = __expf(0.5f * tanhf(c));     // ATTN_SCALE = 0.5
}

__global__ void final_kernel(const float* __restrict__ lw, const float* __restrict__ meanLw,
                             float* __restrict__ outLW, int n) {
  int i = blockIdx.x * blockDim.x + threadIdx.x;
  if (i >= n) return;
  float denom = fmaxf(meanLw[0], 1e-6f);
  outLW[i] = lw[i] / denom;
}

// ---------------------------------------------------------------------------
extern "C" void kernel_launch(void* const* d_in, const int* in_sizes, int n_in,
                              void* d_out, int out_size, void* d_ws, size_t ws_size,
                              hipStream_t stream) {
  const float* obs   = (const float*)d_in[0];
  const float* W1    = (const float*)d_in[1];
  const float* b1    = (const float*)d_in[2];
  const float* W2    = (const float*)d_in[3];
  const float* b2    = (const float*)d_in[4];
  const float* Wr    = (const float*)d_in[5];
  const float* br    = (const float*)d_in[6];
  const float* Win   = (const float*)d_in[7];
  const float* bin_  = (const float*)d_in[8];
  const float* Wout  = (const float*)d_in[9];
  const float* bout  = (const float*)d_in[10];
  const float* gamma = (const float*)d_in[11];
  const float* beta  = (const float*)d_in[12];
  const float* Wh1   = (const float*)d_in[13];
  const float* bh1   = (const float*)d_in[14];
  const float* Wh2   = (const float*)d_in[15];
  const float* bh2   = (const float*)d_in[16];
  float* out = (float*)d_out;

  _Float16* pack = (_Float16*)d_ws;                       // 462848 B used
  float* raw  = (float*)((char*)d_ws + WS_RAW);           // B floats
  float* lw   = (float*)((char*)d_ws + WS_LW);            // B floats
  float* part = (float*)((char*)d_ws + WS_PART);          // 256 partials
  float* scal = part + 256;                               // [0]=mean_raw, [1]=mean_lw

  // 0) pack weights -> f16 WMMA B-fragments (452 tiles x 32 lanes)
  pack_weights<<<(452 * 32 + 255) / 256, 256, 0, stream>>>(W1, W2, Wr, Win, Wout, pack);

  // 1) fused main pipeline: 4096 blocks x 64 threads (2 waves), 32 rows/block
  rca_main<<<B_TOT / 32, 64, 2 * WAVE_LDS, stream>>>(
      obs, pack, b1, b2, br, bin_, bout, gamma, beta, Wh1, bh1, Wh2, bh2, out, raw);

  // 2) mean(raw) over batch (deterministic two-stage tree)
  reduce_partial<<<256, 256, 0, stream>>>(raw, part, B_TOT);
  reduce_final<<<1, 256, 0, stream>>>(part, scal + 0, 1.0f / (float)B_TOT);

  // 3) lw = exp(0.5*tanh(raw - mean)); then mean(lw)
  lw_kernel<<<B_TOT / 256, 256, 0, stream>>>(raw, scal + 0, lw, B_TOT);
  reduce_partial<<<256, 256, 0, stream>>>(lw, part, B_TOT);
  reduce_final<<<1, 256, 0, stream>>>(part, scal + 1, 1.0f / (float)B_TOT);

  // 4) loss_weight = lw / max(mean_lw, 1e-6) -> out[B*64 .. B*65)
  final_kernel<<<B_TOT / 256, 256, 0, stream>>>(lw, scal + 1, out + (size_t)B_TOT * 64, B_TOT);
}
